// lidar_img_encoder_14894946583286
// MI455X (gfx1250) — matrix operand order repaired and checked by reading, not compiled, not run on hardware
//
#include <hip/hip_runtime.h>
#include <hip/hip_bf16.h>
#include <math.h>

// ---------------------------------------------------------------------------
// LiDAR + image encoder for MI455X (gfx1250, wave32, WMMA).
//
// Outputs (flat, concatenated):
//   out0 canvas_ds     [6][112][200]            = 134,400 f32
//   out1 canvas_depth  [6][2][112][112][200]    = 30,105,600 f32
//   out2 img_fore      [6][112][200]            = 134,400 f32
//   out3 max_depth     [6][112][200]            = 134,400 f32
// Workspace: h1 intermediate [6][32][112][200] f32 = 17.2 MB in d_ws.
// ---------------------------------------------------------------------------

typedef __attribute__((ext_vector_type(16))) _Float16 v16h;
typedef __attribute__((ext_vector_type(8)))  float    v8f;

#define NCAM   6
#define H_IMG  112
#define W_IMG  200
#define NPIX   (H_IMG * W_IMG)          // 22400
#define DHW    (112 * NPIX)             // 2,508,800 (D*H*W)
#define OUT0_OFF 0
#define OUT1_OFF 134400
#define OUT2_OFF 30240000
#define OUT3_OFF 30374400
#define TILES_X 13                      // ceil(200/16)
#define CONV1_WAVES 4

// ---------------------------------------------------------------------------
// 1) Project 1M points into 6 cameras; scatter z into canvas_ds (only pixels
//    with x%8==0 && y%8==0 survive the reference's [::8,::8] slice) and the
//    two point features into canvas_depth bins. Plain stores match JAX's
//    unspecified duplicate-scatter semantics.
// ---------------------------------------------------------------------------
__global__ __launch_bounds__(256) void scatter_kernel(
    const float* __restrict__ points, const float* __restrict__ L,
    float* __restrict__ canvas_ds, float* __restrict__ canvas_depth, int N)
{
    __shared__ float Ls[NCAM * 16];
    if (threadIdx.x < NCAM * 16) Ls[threadIdx.x] = L[threadIdx.x];
    __syncthreads();

    int i = blockIdx.x * blockDim.x + threadIdx.x;
    if (i >= N) return;

    // gfx1250 speculative prefetch of the next block's point rows
    __builtin_prefetch(points + (size_t)(i + 256) * 5, 0, 0);

    const float X  = points[(size_t)i * 5 + 0];
    const float Y  = points[(size_t)i * 5 + 1];
    const float Z  = points[(size_t)i * 5 + 2];
    const float f1 = points[(size_t)i * 5 + 3];
    const float f2 = points[(size_t)i * 5 + 4];

    #pragma unroll
    for (int cam = 0; cam < NCAM; ++cam) {
        const float* Lc = &Ls[cam * 16];
        float p0 = Lc[0] * X + Lc[1] * Y + Lc[2]  * Z + Lc[3];
        float p1 = Lc[4] * X + Lc[5] * Y + Lc[6]  * Z + Lc[7];
        float p2 = Lc[8] * X + Lc[9] * Y + Lc[10] * Z + Lc[11];
        float z  = fminf(fmaxf(p2, 1e-4f), 1e4f);
        float px = p0 / z;
        float py = p1 / z;
        bool mask = (px > 0.f) && (px < 1600.f) && (py > 0.f) && (py < 896.f);
        if (!mask) continue;

        int xi = (int)px;   // truncation == floor for positive
        int yi = (int)py;
        if (((xi & 7) == 0) && ((yi & 7) == 0)) {
            canvas_ds[((size_t)cam * H_IMG + (yi >> 3)) * W_IMG + (xi >> 3)] = z;
        }
        int xs = (int)(px * 0.125f);                 // floor(px/8), 0..199
        int ys = (int)(py * 0.125f);                 // floor(py/8), 0..111
        int db = min((int)(z * 1.6f), 111);
        size_t base = (size_t)cam * 2 * DHW + (size_t)db * NPIX + (size_t)ys * W_IMG + xs;
        canvas_depth[base]       = f1;
        canvas_depth[base + DHW] = f2;
    }
}

// ---------------------------------------------------------------------------
// 2) conv1 (4->32, 3x3 SAME) + BN1 as implicit GEMM with WMMA f16.
//    GEMM: D[32, pixels] = W[32, K=36 pad 64] x im2col[K, pixels].
//    One wave32 per 16-pixel strip; LDS patch [4ch][3rows][18cols] with halo.
//    Fragments follow the CDNA5 16-bit layouts (CDNA5 ISA §7.12.2):
//      A: M = lane&15; VGPR v holds K = (v>>2)*16 + (lane>>4)*8 + (v&3)*2 (+1)
//      B: N = lane&15; VGPR v holds K = (v>>2)*16 + (lane>>4)*8 + (v&3)*2 (+1)
//      D: N = lane&15; VGPR r holds M = r + 8*(lane>>4)
//    BN folded to scale/bias on the f32 accumulators.
// ---------------------------------------------------------------------------
__global__ __launch_bounds__(CONV1_WAVES * 32) void conv1_wmma_kernel(
    const float* __restrict__ img, const float* __restrict__ canvas_ds,
    const float* __restrict__ w1,
    const float* __restrict__ g1, const float* __restrict__ b1,
    const float* __restrict__ m1, const float* __restrict__ v1,
    float* __restrict__ h1)
{
    __shared__ float patch[CONV1_WAVES][4][3][18];

    const int lane = threadIdx.x & 31;
    const int wave = threadIdx.x >> 5;
    const int tile = blockIdx.x * CONV1_WAVES + wave;   // grid sized exactly
    const int cam   = tile / (H_IMG * TILES_X);
    const int rem   = tile % (H_IMG * TILES_X);
    const int y     = rem / TILES_X;
    const int xbase = (rem % TILES_X) * 16;

    // Cooperative halo-patch load: 4 ch x 3 rows x 18 cols, OOB -> 0
    for (int idx = lane; idx < 4 * 3 * 18; idx += 32) {
        int ic = idx / 54, r = idx % 54, ry = r / 18, cx = r % 18;
        int yy = y - 1 + ry, xx = xbase - 1 + cx;
        float val = 0.f;
        if (yy >= 0 && yy < H_IMG && xx >= 0 && xx < W_IMG) {
            val = (ic < 3) ? img[(((size_t)cam * 3 + ic) * H_IMG + yy) * W_IMG + xx]
                           : canvas_ds[((size_t)cam * H_IMG + yy) * W_IMG + xx];
        }
        patch[wave][ic][ry][cx] = val;
    }
    __syncthreads();

    const int Mrow = lane & 15;
    const int hsel = lane >> 4;
    const int Ncol = lane & 15;

    // A fragments: weights (constant across tiles), 2 M-tiles x 2 K-steps
    v16h a[2][2];
    #pragma unroll
    for (int m = 0; m < 2; ++m) {
        #pragma unroll
        for (int s = 0; s < 2; ++s) {
            #pragma unroll
            for (int v = 0; v < 8; ++v) {
                int klo = s * 32 + (v >> 2) * 16 + hsel * 8 + (v & 3) * 2;
                int oc  = m * 16 + Mrow;
                float e0 = (klo     < 36) ? w1[oc * 36 + klo]     : 0.f;
                float e1 = (klo + 1 < 36) ? w1[oc * 36 + klo + 1] : 0.f;
                a[m][s][2 * v]     = (_Float16)e0;
                a[m][s][2 * v + 1] = (_Float16)e1;
            }
        }
    }

    // B fragments: im2col of the LDS patch, K = ic*9 + ky*3 + kx (pad to 64)
    v16h bfr[2];
    #pragma unroll
    for (int s = 0; s < 2; ++s) {
        #pragma unroll
        for (int v = 0; v < 8; ++v) {
            int klo = s * 32 + (v >> 2) * 16 + hsel * 8 + (v & 3) * 2;
            #pragma unroll
            for (int e = 0; e < 2; ++e) {
                int k = klo + e;
                float f = 0.f;
                if (k < 36) {
                    int ic = k / 9, rr = k % 9, ky = rr / 3, kx = rr % 3;
                    f = patch[wave][ic][ky][Ncol + kx];
                }
                bfr[s][2 * v + e] = (_Float16)f;
            }
        }
    }

    v8f acc0 = {};
    v8f acc1 = {};
    acc0 = __builtin_amdgcn_wmma_f32_16x16x32_f16(false, a[0][0], false, bfr[0], (short)0, acc0, false, false);
    acc0 = __builtin_amdgcn_wmma_f32_16x16x32_f16(false, a[0][1], false, bfr[1], (short)0, acc0, false, false);
    acc1 = __builtin_amdgcn_wmma_f32_16x16x32_f16(false, a[1][0], false, bfr[0], (short)0, acc1, false, false);
    acc1 = __builtin_amdgcn_wmma_f32_16x16x32_f16(false, a[1][1], false, bfr[1], (short)0, acc1, false, false);

    // BN1 fold + store h1 (workspace)
    const int x = xbase + Ncol;
    if (x < W_IMG) {
        #pragma unroll
        for (int r = 0; r < 8; ++r) {
            int Mr = r + 8 * hsel;
            int oc = Mr;
            float sc = g1[oc] * rsqrtf(v1[oc] + 1e-3f);
            float bb = b1[oc] - m1[oc] * sc;
            h1[(((size_t)cam * 32 + oc) * H_IMG + y) * W_IMG + x] = acc0[r] * sc + bb;
            oc = 16 + Mr;
            sc = g1[oc] * rsqrtf(v1[oc] + 1e-3f);
            bb = b1[oc] - m1[oc] * sc;
            h1[(((size_t)cam * 32 + oc) * H_IMG + y) * W_IMG + x] = acc1[r] * sc + bb;
        }
    }
}

// ---------------------------------------------------------------------------
// 3) conv2 (32->1, 3x3 SAME) + BN2 + sigmoid. M=1 would waste 15/16 of a
//    WMMA tile, so plain VALU (288 MACs/pixel, L2-resident reads).
// ---------------------------------------------------------------------------
__global__ __launch_bounds__(256) void conv2_kernel(
    const float* __restrict__ h1, const float* __restrict__ w2,
    const float* __restrict__ g2, const float* __restrict__ b2,
    const float* __restrict__ m2, const float* __restrict__ v2,
    float* __restrict__ out2)
{
    int p = blockIdx.x * 256 + threadIdx.x;
    if (p >= NCAM * NPIX) return;
    int cam = p / NPIX, rem = p % NPIX;
    int y = rem / W_IMG, x = rem % W_IMG;

    float acc = 0.f;
    for (int ic = 0; ic < 32; ++ic) {
        const float* hp = h1 + ((size_t)cam * 32 + ic) * (size_t)NPIX;
        #pragma unroll
        for (int ky = 0; ky < 3; ++ky) {
            int yy = y + ky - 1;
            if (yy < 0 || yy >= H_IMG) continue;
            #pragma unroll
            for (int kx = 0; kx < 3; ++kx) {
                int xx = x + kx - 1;
                if (xx < 0 || xx >= W_IMG) continue;
                acc += w2[ic * 9 + ky * 3 + kx] * hp[yy * W_IMG + xx];
            }
        }
    }
    float sc = g2[0] * rsqrtf(v2[0] + 1e-3f);
    float h  = acc * sc + (b2[0] - m2[0] * sc);
    out2[p] = 1.f / (1.f + expf(-h));
}

// ---------------------------------------------------------------------------
// 4) 5x5 max-pool on canvas_ds, zero-padded by 2 (matches pad-0 + reduce
//    window with -inf init).
// ---------------------------------------------------------------------------
__global__ __launch_bounds__(256) void maxpool_kernel(
    const float* __restrict__ canvas_ds, float* __restrict__ out3)
{
    int p = blockIdx.x * 256 + threadIdx.x;
    if (p >= NCAM * NPIX) return;
    int cam = p / NPIX, rem = p % NPIX;
    int y = rem / W_IMG, x = rem % W_IMG;

    float m = -INFINITY;
    #pragma unroll
    for (int dy = -2; dy <= 2; ++dy) {
        int yy = y + dy;
        #pragma unroll
        for (int dx = -2; dx <= 2; ++dx) {
            int xx = x + dx;
            float v = 0.f;   // zero padding
            if (yy >= 0 && yy < H_IMG && xx >= 0 && xx < W_IMG)
                v = canvas_ds[((size_t)cam * H_IMG + yy) * W_IMG + xx];
            m = fmaxf(m, v);
        }
    }
    out3[p] = m;
}

// ---------------------------------------------------------------------------
extern "C" void kernel_launch(void* const* d_in, const int* in_sizes, int n_in,
                              void* d_out, int out_size, void* d_ws, size_t ws_size,
                              hipStream_t stream) {
    const float* points = (const float*)d_in[0];
    const float* L      = (const float*)d_in[1];
    const float* img    = (const float*)d_in[2];
    const float* w1     = (const float*)d_in[3];
    const float* g1     = (const float*)d_in[4];
    const float* b1     = (const float*)d_in[5];
    const float* m1     = (const float*)d_in[6];
    const float* v1     = (const float*)d_in[7];
    const float* w2     = (const float*)d_in[8];
    const float* g2     = (const float*)d_in[9];
    const float* b2     = (const float*)d_in[10];
    const float* m2     = (const float*)d_in[11];
    const float* v2     = (const float*)d_in[12];

    float* out  = (float*)d_out;
    float* out0 = out + OUT0_OFF;   // canvas_ds
    float* out1 = out + OUT1_OFF;   // canvas_depth
    float* out2 = out + OUT2_OFF;   // img_fore
    float* out3 = out + OUT3_OFF;   // max_depth
    float* h1   = (float*)d_ws;     // needs 6*32*112*200*4 = 17.2 MB

    const int N = in_sizes[0] / 5;  // 1,000,000 points

    // Zero the scatter targets (canvas_ds + canvas_depth are contiguous).
    hipMemsetAsync(out, 0, (size_t)(OUT1_OFF + NCAM * 2 * (size_t)DHW) * sizeof(float), stream);

    scatter_kernel<<<(N + 255) / 256, 256, 0, stream>>>(points, L, out0, out1, N);

    conv1_wmma_kernel<<<(NCAM * H_IMG * TILES_X) / CONV1_WAVES, CONV1_WAVES * 32, 0, stream>>>(
        img, out0, w1, g1, b1, m1, v1, h1);

    conv2_kernel<<<(NCAM * NPIX + 255) / 256, 256, 0, stream>>>(h1, w2, g2, b2, m2, v2, out2);

    maxpool_kernel<<<(NCAM * NPIX + 255) / 256, 256, 0, stream>>>(out0, out3);
}
